// EEG_GAT_Model_44641890075103
// MI455X (gfx1250) — matrix operand order repaired and checked by reading, not compiled, not run on hardware
//
#include <hip/hip_runtime.h>
#include <hip/hip_bf16.h>
#include <math.h>

typedef __attribute__((ext_vector_type(16))) __bf16        v16bf;
typedef __attribute__((ext_vector_type(8)))  float         v8f;
typedef __attribute__((ext_vector_type(8)))  unsigned int  v8u;
typedef __attribute__((ext_vector_type(4)))  int           v4i;

#define NND 19            // electrodes / graph nodes
#define DHID 512          // H*C
#define NHEAD 8
#define CHEAD 64

#define TILE_M 32
#define TILE_N 256
#define TILE_K 32
#define A_PAD 8           // As row = 40 ushorts = 80B: 16B-aligned, distinct bank residues

#define AS1 __attribute__((address_space(1)))
#define AS3 __attribute__((address_space(3)))

#if defined(__has_builtin)
#if __has_builtin(__builtin_amdgcn_global_load_async_to_lds_b128)
#define USE_ASYNC_LDS 1
#endif
#endif

// ---------- helpers ----------
__device__ __forceinline__ unsigned short f32_to_bf16_rne(float f) {
  unsigned int u = __builtin_bit_cast(unsigned int, f);
  u = (u + 0x7FFFu + ((u >> 16) & 1u)) >> 16;
  return (unsigned short)u;
}

__device__ __forceinline__ void wait_async_cnt0() {
#if defined(__has_builtin)
#if __has_builtin(__builtin_amdgcn_s_wait_asynccnt)
  __builtin_amdgcn_s_wait_asynccnt(0);
#else
  asm volatile("s_wait_asynccnt 0x0" ::: "memory");
#endif
#else
  asm volatile("s_wait_asynccnt 0x0" ::: "memory");
#endif
}

// ---------- kernel 0: x[B,S,N](f32) -> XT[B,N,S](bf16) ----------
__global__ __launch_bounds__(256) void transpose_x(const float* __restrict__ x,
                                                   unsigned short* __restrict__ xt) {
  const size_t idx = (size_t)blockIdx.x * 256 + threadIdx.x;  // over B*19*256
  const int s  = (int)(idx & 255);
  const int bn = (int)(idx >> 8);       // b*19 + n
  const int n  = bn % NND;
  const int b  = bn / NND;
  xt[idx] = f32_to_bf16_rne(x[((size_t)b * 256 + s) * NND + n]);
}

// ---------- kernel 1: f32 -> bf16 weight conversion ----------
__global__ __launch_bounds__(256) void cvt_f32_bf16(const float* __restrict__ in,
                                                    unsigned short* __restrict__ out,
                                                    int n) {
  const int i = blockIdx.x * 256 + threadIdx.x;
  if (i < n) out[i] = f32_to_bf16_rne(in[i]);
}

// ---------- kernel 2: C[M,N](f32) = A[M,K](bf16) @ W[K,N](bf16) + bias[N] ----------
// 8 waves: 2 (M) x 4 (N); each wave: 16x64 strip = 4 x v_wmma_f32_16x16x32_bf16 / K-step.
// A tile staged via GLOBAL_LOAD_ASYNC_TO_LDS_B128 (ASYNCcnt), W tile transposed manually.
__global__ __launch_bounds__(256)
void gemm_bf16_wmma(const unsigned short* __restrict__ A,
                    const unsigned short* __restrict__ Wb,
                    const float* __restrict__ bias, float* __restrict__ Cout,
                    int M, int K, int N) {
  __shared__ unsigned short As[TILE_M][TILE_K + A_PAD];  // 80B rows
  __shared__ unsigned short Bs[TILE_N][TILE_K + 4];      // transposed W tile, 18-dword stride

  const int tid  = threadIdx.x;
  const int lane = tid & 31;
  const int wave = tid >> 5;
  const int mw   = wave >> 2;        // 0..1
  const int nw   = wave & 3;         // 0..3
  const int half = lane >> 4;
  const int l16  = lane & 15;
  const int gm0  = blockIdx.x * TILE_M;
  const int gn0  = blockIdx.y * TILE_N;

  v8f acc[4] = {};

  const int arow = tid >> 2;          // 0..63 (use tid<128 -> rows 0..31)
  const int acol = (tid & 3) << 3;    // bf16 offset within row, 16B chunks
  const int bn8  = (tid & 31) << 3;   // B tile col group (8 bf16)
  const int bk0  = tid >> 5;          // B tile base row 0..7

  for (int k0 = 0; k0 < K; k0 += TILE_K) {
    // --- stage A tile: 32 rows x 64B, 128 lanes x 16B async copies ---
    if (tid < 128) {
      const unsigned short* gp = A + (size_t)(gm0 + arow) * K + k0 + acol;
#ifdef USE_ASYNC_LDS
      __builtin_amdgcn_global_load_async_to_lds_b128(
          (AS1 v4i*)gp, (AS3 v4i*)&As[arow][acol], 0, 0);
#else
      *(uint4*)&As[arow][acol] = *(const uint4*)gp;
#endif
    }
    __builtin_prefetch(A + (size_t)(gm0 + (tid >> 3)) * K + k0 + TILE_K, 0, 3);

    // --- stage W tile, transposed into LDS: Bs[n][k] ---
#pragma unroll
    for (int p = 0; p < 4; ++p) {
      const int kk = bk0 + (p << 3);
      const uint4 wv = *(const uint4*)(Wb + (size_t)(k0 + kk) * N + gn0 + bn8);
      Bs[bn8 + 0][kk] = (unsigned short)(wv.x);
      Bs[bn8 + 1][kk] = (unsigned short)(wv.x >> 16);
      Bs[bn8 + 2][kk] = (unsigned short)(wv.y);
      Bs[bn8 + 3][kk] = (unsigned short)(wv.y >> 16);
      Bs[bn8 + 4][kk] = (unsigned short)(wv.z);
      Bs[bn8 + 5][kk] = (unsigned short)(wv.z >> 16);
      Bs[bn8 + 6][kk] = (unsigned short)(wv.w);
      Bs[bn8 + 7][kk] = (unsigned short)(wv.w >> 16);
    }
    __builtin_prefetch(Wb + (size_t)(k0 + TILE_K + bk0) * N + gn0 + bn8, 0, 3);
#ifdef USE_ASYNC_LDS
    wait_async_cnt0();
#endif
    __syncthreads();

    // --- A fragment (ISA 16-bit A 16x32 layout: K groups of 8 per lane-half) ---
    v8u araw;
#pragma unroll
    for (int v = 0; v < 8; ++v) {
      const int kb = ((v < 4) ? (v << 1) : (16 + ((v - 4) << 1))) + (half << 3);
      araw[v] = *(const unsigned int*)&As[mw * 16 + l16][kb];
    }
    const v16bf afrag = __builtin_bit_cast(v16bf, araw);

    // --- 4 B fragments + WMMA (B layout: same K groups of 8 per lane-half as A) ---
#pragma unroll
    for (int f = 0; f < 4; ++f) {
      const int ncol = nw * 64 + f * 16 + l16;
      v8u braw;
#pragma unroll
      for (int v = 0; v < 8; ++v) {
        const int kb = ((v < 4) ? (v << 1) : (16 + ((v - 4) << 1))) + (half << 3);
        braw[v] = *(const unsigned int*)&Bs[ncol][kb];
      }
      const v16bf bfrag = __builtin_bit_cast(v16bf, braw);
      acc[f] = __builtin_amdgcn_wmma_f32_16x16x32_bf16(
          false, afrag, false, bfrag, (short)0, acc[f], false, false);
    }
    __syncthreads();
  }

  // --- store D (C/D layout: lane = N, VGPR r -> M = r + 8*half) + bias ---
#pragma unroll
  for (int f = 0; f < 4; ++f) {
    const int n  = gn0 + nw * 64 + f * 16 + l16;
    const float bv = bias[n];
#pragma unroll
    for (int r = 0; r < 8; ++r) {
      const int m = gm0 + mw * 16 + r + (half << 3);
      Cout[(size_t)m * N + n] = acc[f][r] + bv;
    }
  }
}

// ---------- kernel 3: GATv2 attention per (batch, head) ----------
// Writes bf16 (layer-2 GEMM A operand) and/or f32 (pool input).
__global__ __launch_bounds__(256)
void gat_attention(const float* __restrict__ XL, const float* __restrict__ XR,
                   const float* __restrict__ att, const float* __restrict__ bias,
                   float* __restrict__ HoutF, unsigned short* __restrict__ HoutBf,
                   int apply_elu) {
  const int b = blockIdx.x >> 3;
  const int h = blockIdx.x & 7;
  __shared__ float xl_s[NND][CHEAD];
  __shared__ float xr_s[NND][CHEAD];
  __shared__ float att_s[CHEAD];
  __shared__ float es[NND][NND + 1];
  const int tid = threadIdx.x;

  if (tid < CHEAD) att_s[tid] = att[h * CHEAD + tid];
  for (int idx = tid; idx < NND * CHEAD; idx += 256) {
    const int j = idx >> 6, c = idx & 63;
    const size_t off = ((size_t)(b * NND + j) * DHID) + h * CHEAD + c;
    xl_s[j][c] = XL[off];
    xr_s[j][c] = XR[off];
  }
  __syncthreads();

  // logits: e[i][j] = att_h . leaky_relu(xr_i + xl_j)
  for (int p = tid; p < NND * NND; p += 256) {
    const int i = p / NND, j = p % NND;
    float s = 0.f;
#pragma unroll
    for (int c = 0; c < CHEAD; ++c) {
      float z = xr_s[i][c] + xl_s[j][c];
      z = (z > 0.f) ? z : 0.2f * z;
      s = fmaf(att_s[c], z, s);
    }
    es[i][j] = s;
  }
  __syncthreads();

  // softmax over sources j
  if (tid < NND) {
    const int i = tid;
    float mx = -3.4e38f;
    for (int j = 0; j < NND; ++j) mx = fmaxf(mx, es[i][j]);
    float sum = 0.f;
    for (int j = 0; j < NND; ++j) { const float e = __expf(es[i][j] - mx); es[i][j] = e; sum += e; }
    const float inv = 1.f / sum;
    for (int j = 0; j < NND; ++j) es[i][j] *= inv;
  }
  __syncthreads();

  // aggregate messages (+bias, optional ELU)
  for (int idx = tid; idx < NND * CHEAD; idx += 256) {
    const int i = idx >> 6, c = idx & 63;
    float s = 0.f;
#pragma unroll
    for (int j = 0; j < NND; ++j) s = fmaf(es[i][j], xl_s[j][c], s);
    s += bias[h * CHEAD + c];
    if (apply_elu) s = (s > 0.f) ? s : expm1f(s);
    const size_t off = ((size_t)(b * NND + i) * DHID) + h * CHEAD + c;
    if (HoutBf) HoutBf[off] = f32_to_bf16_rne(s);
    if (HoutF)  HoutF[off]  = s;
  }
}

// ---------- kernel 4: attention pooling over nodes + final fc ----------
__global__ __launch_bounds__(256)
void pool_fc(const float* __restrict__ H2, const float* __restrict__ poolW,
             const float* __restrict__ poolB, const float* __restrict__ fcW,
             const float* __restrict__ fcB, float* __restrict__ out) {
  const int b = blockIdx.x;
  __shared__ float sc[NND];
  __shared__ float red[256];
  const int tid = threadIdx.x;
  const float* hb = H2 + (size_t)b * NND * DHID;

  if (tid < NND) {
    float s = 0.f;
    for (int c = 0; c < DHID; ++c) s = fmaf(hb[tid * DHID + c], poolW[c], s);
    sc[tid] = s + poolB[0];
  }
  __syncthreads();
  if (tid == 0) {
    float mx = sc[0];
    for (int n = 1; n < NND; ++n) mx = fmaxf(mx, sc[n]);
    float sum = 0.f;
    for (int n = 0; n < NND; ++n) { const float e = __expf(sc[n] - mx); sc[n] = e; sum += e; }
    const float inv = 1.f / sum;
    for (int n = 0; n < NND; ++n) sc[n] *= inv;
  }
  __syncthreads();

  float part = 0.f;
  for (int c = tid; c < DHID; c += 256) {
    float p = 0.f;
#pragma unroll
    for (int n = 0; n < NND; ++n) p = fmaf(sc[n], hb[n * DHID + c], p);
    part = fmaf(p, fcW[c], part);
  }
  red[tid] = part;
  __syncthreads();
  for (int s = 128; s > 0; s >>= 1) {
    if (tid < s) red[tid] += red[tid + s];
    __syncthreads();
  }
  if (tid == 0) out[b] = red[0] + fcB[0];
}

// ---------- host launcher ----------
extern "C" void kernel_launch(void* const* d_in, const int* in_sizes, int n_in,
                              void* d_out, int out_size, void* d_ws, size_t ws_size,
                              hipStream_t stream) {
  (void)in_sizes; (void)n_in; (void)out_size; (void)ws_size;
  const int B = 2048, S = 256;
  const int M = B * NND;                 // 38912 rows, multiple of TILE_M

  const float* x       = (const float*)d_in[0];
  const float* g1_Wl   = (const float*)d_in[1];
  const float* g1_bl   = (const float*)d_in[2];
  const float* g1_Wr   = (const float*)d_in[3];
  const float* g1_br   = (const float*)d_in[4];
  const float* g1_att  = (const float*)d_in[5];
  const float* g1_bias = (const float*)d_in[6];
  const float* g2_Wl   = (const float*)d_in[7];
  const float* g2_bl   = (const float*)d_in[8];
  const float* g2_Wr   = (const float*)d_in[9];
  const float* g2_br   = (const float*)d_in[10];
  const float* g2_att  = (const float*)d_in[11];
  const float* g2_bias = (const float*)d_in[12];
  const float* pool_W  = (const float*)d_in[13];
  const float* pool_b  = (const float*)d_in[14];
  const float* fc_W    = (const float*)d_in[15];
  const float* fc_b    = (const float*)d_in[16];

  char* ws = (char*)d_ws;
  size_t off = 0;
  auto alloc = [&](size_t bytes) -> void* {
    void* p = ws + off;
    off = (off + bytes + 255) & ~(size_t)255;
    return p;
  };

  unsigned short* XTb  = (unsigned short*)alloc((size_t)M * S * 2);      // [B,19,256] bf16
  float*          XL   = (float*)alloc((size_t)M * DHID * sizeof(float));
  float*          XR   = (float*)alloc((size_t)M * DHID * sizeof(float));
  unsigned short* H1b  = (unsigned short*)alloc((size_t)M * DHID * 2);   // layer-2 A (bf16)
  float*          H2f  = (float*)alloc((size_t)M * DHID * sizeof(float));
  unsigned short* w1l  = (unsigned short*)alloc((size_t)S * DHID * 2);
  unsigned short* w1r  = (unsigned short*)alloc((size_t)S * DHID * 2);
  unsigned short* w2l  = (unsigned short*)alloc((size_t)DHID * DHID * 2);
  unsigned short* w2r  = (unsigned short*)alloc((size_t)DHID * DHID * 2);

  // 0) transpose + convert x -> XTb
  transpose_x<<<(M * S) / 256, 256, 0, stream>>>(x, XTb);

  // 1) weights to bf16
  const int n1 = S * DHID, n2 = DHID * DHID;
  cvt_f32_bf16<<<(n1 + 255) / 256, 256, 0, stream>>>(g1_Wl, w1l, n1);
  cvt_f32_bf16<<<(n1 + 255) / 256, 256, 0, stream>>>(g1_Wr, w1r, n1);
  cvt_f32_bf16<<<(n2 + 255) / 256, 256, 0, stream>>>(g2_Wl, w2l, n2);
  cvt_f32_bf16<<<(n2 + 255) / 256, 256, 0, stream>>>(g2_Wr, w2r, n2);

  const dim3 ggrid(M / TILE_M, DHID / TILE_N);   // 1216 x 2

  // 2) layer-1 GEMMs + attention (+ELU), attention emits bf16 H1
  gemm_bf16_wmma<<<ggrid, 256, 0, stream>>>(XTb, w1l, g1_bl, XL, M, S, DHID);
  gemm_bf16_wmma<<<ggrid, 256, 0, stream>>>(XTb, w1r, g1_br, XR, M, S, DHID);
  gat_attention<<<B * NHEAD, 256, 0, stream>>>(XL, XR, g1_att, g1_bias,
                                               nullptr, H1b, 1);

  // 3) layer-2 GEMMs + attention (f32 H2 for pooling)
  gemm_bf16_wmma<<<ggrid, 256, 0, stream>>>(H1b, w2l, g2_bl, XL, M, DHID, DHID);
  gemm_bf16_wmma<<<ggrid, 256, 0, stream>>>(H1b, w2r, g2_br, XR, M, DHID, DHID);
  gat_attention<<<B * NHEAD, 256, 0, stream>>>(XL, XR, g2_att, g2_bias,
                                               H2f, nullptr, 0);

  // 4) pool + fc -> logits [B]
  pool_fc<<<B, 256, 0, stream>>>(H2f, pool_W, pool_b, fc_W, fc_b, (float*)d_out);
}
